// GvHeat_81612968559187
// MI455X (gfx1250) — matrix operand order repaired and checked by reading, not compiled, not run on hardware
//
#include <hip/hip_runtime.h>
#include <hip/hip_bf16.h>
#include <math.h>

// ---------------------------------------------------------------------------
// Problem constants (from reference): B=8, C=96, H=W=128
// ---------------------------------------------------------------------------
#define BB   8
#define CC   96
#define C2   192
#define HH   128
#define WW   128
#define HWN  (HH * WW)          // 16384
#define BHW  (BB * HWN)         // 131072

typedef __attribute__((ext_vector_type(16))) __bf16 v16bf;
typedef __attribute__((ext_vector_type(8)))  float  v8f;

// ---------------------------------------------------------------------------
// Batched GEMM:  C[m,n] = sum_k A[m,k] * B[k,n] (+bias[n]) (ReLU?)
// REQUIRES: M % 128 == 0, N % 32 == 0, K % 32 == 0 (true for every call here).
//   A: fp32 row-major (lda, batch strideA; 0 => shared)
//   B: fp32; OPB==0: B[k*ldb+n] (activation slabs)  OPB==1: B[n*ldb+k] (weights [N,K])
// bf16 LDS staging (double-buffered), fp32 accumulate via v_wmma_f32_16x16x32_bf16.
// Block: 128 threads = 4 waves; block tile 128x32; wave tile 32x32; K-step 32.
// ---------------------------------------------------------------------------
#define TM 128
#define TN 32
#define TK 32
#define KPAD 4
#define KROW (TK + KPAD)        // 36 bf16 elements per LDS row
#define ROWU (KROW / 2)         // 18 dwords per LDS row

template<int OPB>
__global__ __launch_bounds__(128)
void gemm_bf16_wmma(const float* __restrict__ A, long long lda, long long strideA,
                    const float* __restrict__ Bm, long long ldb, long long strideB,
                    float* __restrict__ Cm, long long ldc, long long strideC,
                    const float* __restrict__ bias, int relu, int K)
{
    __shared__ __bf16 As[2][TM][KROW];
    __shared__ __bf16 Bs[2][TN][KROW];

    const int tid = threadIdx.x;
    const long long zb = blockIdx.z;
    A  += zb * strideA;
    Bm += zb * strideB;
    Cm += zb * strideC;

    const int m0 = blockIdx.y * TM;
    const int n0 = blockIdx.x * TN;

    const int wave = tid >> 5;       // 0..3, stacked along M
    const int lane = tid & 31;
    const int lrow = lane & 15;
    const int hlf  = lane >> 4;

    v8f acc[2][2] = {};
    union Frag  { v16bf v; unsigned u[8]; };
    union Pack4 { __bf16 h[4]; unsigned long long u; };

    // ---- branch-free tile staging (all shapes divide the tiles exactly) ----
    auto stageA = [&](int buf, int k0) {
        #pragma unroll
        for (int i = 0; i < 8; ++i) {
            int idx = tid + i * 128;          // 1024 float4s: 128 rows x 8 quads
            int r = idx >> 3, q = idx & 7;
            const float4 f = *(const float4*)(A + (long long)(m0 + r) * lda + k0 + q * 4);
            Pack4 pk;
            pk.h[0] = (__bf16)f.x; pk.h[1] = (__bf16)f.y;
            pk.h[2] = (__bf16)f.z; pk.h[3] = (__bf16)f.w;
            *(unsigned long long*)&As[buf][r][q * 4] = pk.u;
        }
    };
    auto stageB = [&](int buf, int k0) {
        if (OPB) {
            // weight [N,K] row-major: contiguous along k
            #pragma unroll
            for (int i = 0; i < 2; ++i) {
                int idx = tid + i * 128;      // 256 float4s: 32 rows x 8 quads
                int n = idx >> 3, q = idx & 7;
                const float4 f = *(const float4*)(Bm + (long long)(n0 + n) * ldb + k0 + q * 4);
                Pack4 pk;
                pk.h[0] = (__bf16)f.x; pk.h[1] = (__bf16)f.y;
                pk.h[2] = (__bf16)f.z; pk.h[3] = (__bf16)f.w;
                *(unsigned long long*)&Bs[buf][n][q * 4] = pk.u;
            }
        } else {
            // activation slab [K,N]: contiguous along n; transpose into Bs[n][k]
            #pragma unroll
            for (int i = 0; i < 2; ++i) {
                int idx = tid + i * 128;      // 256 float4s: 32 k-rows x 8 n-quads
                int kk = idx >> 3, q = idx & 7;
                const float4 f = *(const float4*)(Bm + (long long)(k0 + kk) * ldb + n0 + q * 4);
                Bs[buf][q * 4 + 0][kk] = (__bf16)f.x;
                Bs[buf][q * 4 + 1][kk] = (__bf16)f.y;
                Bs[buf][q * 4 + 2][kk] = (__bf16)f.z;
                Bs[buf][q * 4 + 3][kk] = (__bf16)f.w;
            }
        }
    };

    const int nk = K / TK;
    stageA(0, 0);
    stageB(0, 0);
    __syncthreads();

    for (int kt = 0; kt < nk; ++kt) {
        const int cur = kt & 1;
        if (kt + 1 < nk) {
            stageA(cur ^ 1, (kt + 1) * TK);
            stageB(cur ^ 1, (kt + 1) * TK);
        }
        if (kt + 2 < nk)   // gfx1250 global_prefetch_b8 for the tile after next
            __builtin_prefetch(A + (long long)(m0 + tid) * lda + (kt + 2) * TK, 0, 0);

        // ---- gather WMMA fragments (CDNA5 16-bit A/B VGPR layouts) ----
        const unsigned* As32 = (const unsigned*)&As[cur][0][0];
        const unsigned* Bs32 = (const unsigned*)&Bs[cur][0][0];

        Frag af[2], bfr[2];
        #pragma unroll
        for (int mi = 0; mi < 2; ++mi) {
            int m = wave * 32 + mi * 16 + lrow;
            const unsigned* rp = As32 + m * ROWU;
            #pragma unroll
            for (int j = 0; j < 8; ++j) {
                // A 16x32: vgpr j -> K pair {2j+8h} (j<4) / {16+2(j-4)+8h}
                int p = (j < 4) ? (j + 4 * hlf) : (8 + (j - 4) + 4 * hlf);
                af[mi].u[j] = rp[p];
            }
        }
        #pragma unroll
        for (int ni = 0; ni < 2; ++ni) {
            int n = ni * 16 + lrow;
            const unsigned* rp = Bs32 + n * ROWU;
            #pragma unroll
            for (int j = 0; j < 8; ++j) {
                // B 32x16: vgpr j -> K pair {2j+8h} (j<4) / {16+2(j-4)+8h}
                bfr[ni].u[j] = rp[(j < 4) ? (j + 4 * hlf) : (8 + (j - 4) + 4 * hlf)];
            }
        }

        #pragma unroll
        for (int mi = 0; mi < 2; ++mi)
            #pragma unroll
            for (int ni = 0; ni < 2; ++ni)
                acc[mi][ni] = __builtin_amdgcn_wmma_f32_16x16x32_bf16(
                    false, af[mi].v, false, bfr[ni].v,
                    (short)0, acc[mi][ni], false, false);

        __syncthreads();
    }

    // ---- epilogue: C layout vgpr r -> M = r + 8*hlf, N = lrow (coalesced) ----
    #pragma unroll
    for (int mi = 0; mi < 2; ++mi) {
        #pragma unroll
        for (int ni = 0; ni < 2; ++ni) {
            int gn = n0 + ni * 16 + lrow;
            float bv = bias ? bias[gn] : 0.f;
            #pragma unroll
            for (int r = 0; r < 8; ++r) {
                int gm = m0 + wave * 32 + mi * 16 + r + 8 * hlf;
                float v = acc[mi][ni][r] + bv;
                if (relu) v = v > 0.f ? v : 0.f;
                Cm[(long long)gm * ldc + gn] = v;
            }
        }
    }
}

// ---------------------------------------------------------------------------
// Precompute: DCT matrix cn (and transpose), and e[n,m] = wn^2 + wm^2
// ---------------------------------------------------------------------------
__global__ void precompute_kernel(float* cn, float* cnT, float* e)
{
    int idx = blockIdx.x * blockDim.x + threadIdx.x;
    if (idx >= HH * HH) return;
    int n = idx / HH, h = idx % HH;
    const float PI = 3.14159265358979323846f;
    float v = cosf((float)n * ((h + 0.5f) / (float)HH) * PI) * sqrtf(2.f / (float)HH);
    if (n == 0) v *= 0.70710678118654752440f;
    cn[n * HH + h] = v;
    cnT[h * HH + n] = v;
    float wn = PI * (float)n / (float)HH;
    float wm = PI * (float)h / (float)WW;
    e[idx] = wn * wn + wm * wm;
}

// ---------------------------------------------------------------------------
// Depthwise 3x3 conv (pad 1) NCHW -> NHWC, + bias
// ---------------------------------------------------------------------------
__global__ void dwconv_kernel(const float* __restrict__ x,
                              const float* __restrict__ w9,
                              const float* __restrict__ bias,
                              float* __restrict__ out)
{
    long long idx = (long long)blockIdx.x * blockDim.x + threadIdx.x;
    if (idx >= (long long)BB * CC * HWN) return;
    int wcol = (int)(idx % WW);
    int h    = (int)((idx / WW) % HH);
    int c    = (int)((idx / (long long)HWN) % CC);
    int b    = (int)(idx / ((long long)CC * HWN));
    float s = bias[c];
    const float* xp = x + ((long long)(b * CC + c) * HH) * WW;
    #pragma unroll
    for (int i = 0; i < 3; ++i) {
        int hh = h + i - 1;
        if (hh < 0 || hh >= HH) continue;
        #pragma unroll
        for (int j = 0; j < 3; ++j) {
            int ww2 = wcol + j - 1;
            if (ww2 < 0 || ww2 >= WW) continue;
            s += xp[hh * WW + ww2] * w9[c * 9 + i * 3 + j];
        }
    }
    out[((long long)(b * HWN) + h * WW + wcol) * CC + c] = s;
}

// freq_embed + mean_b(outline)
__global__ void meanfe_kernel(const float* __restrict__ fe,
                              const float* __restrict__ ot,
                              float* __restrict__ out)
{
    long long idx = (long long)blockIdx.x * blockDim.x + threadIdx.x;
    if (idx >= (long long)HWN * CC) return;
    float s = 0.f;
    #pragma unroll
    for (int b = 0; b < BB; ++b) s += ot[(long long)b * HWN * CC + idx];
    out[idx] = fe[idx] + 0.125f * s;
}

// buf[b,n,m,c] *= exp(-k[n,m,c] * e[n,m])   (Cc = 96 or 192)
__global__ void decay_kernel(float* __restrict__ buf,
                             const float* __restrict__ k,
                             const float* __restrict__ e,
                             int Cc, long long total)
{
    long long idx = (long long)blockIdx.x * blockDim.x + threadIdx.x;
    if (idx >= total) return;
    int c = (int)(idx % Cc);
    int m = (int)((idx / Cc) % WW);
    int n = (int)((idx / ((long long)Cc * WW)) % HH);
    buf[idx] *= __expf(-k[((long long)n * WW + m) * Cc + c] * e[n * WW + m]);
}

// cat = [x1 | outline]  -> [BHW, 192]
__global__ void concat_kernel(const float* __restrict__ x1,
                              const float* __restrict__ ot,
                              float* __restrict__ out)
{
    long long idx = (long long)blockIdx.x * blockDim.x + threadIdx.x;
    if (idx >= (long long)BHW * C2) return;
    int d = (int)(idx % C2);
    long long bhw = idx / C2;
    out[idx] = (d < CC) ? x1[bhw * CC + d] : ot[bhw * CC + (d - CC)];
}

// (1/8) * sum_b cat  -> [HW, 192]
__global__ void batchsum_kernel(const float* __restrict__ cat,
                                float* __restrict__ out)
{
    long long idx = (long long)blockIdx.x * blockDim.x + threadIdx.x;
    if (idx >= (long long)HWN * C2) return;
    float s = 0.f;
    #pragma unroll
    for (int b = 0; b < BB; ++b) s += cat[(long long)b * HWN * C2 + idx];
    out[idx] = 0.125f * s;
}

// LayerNorm over C=96 then gate by silu(z); one wave32 per row, in place
__global__ void ln_gate_kernel(float* __restrict__ y,
                               const float* __restrict__ z,
                               const float* __restrict__ g,
                               const float* __restrict__ bta,
                               int rows)
{
    int wid  = (blockIdx.x * blockDim.x + threadIdx.x) >> 5;
    int lane = threadIdx.x & 31;
    if (wid >= rows) return;
    float*       row = y + (long long)wid * CC;
    const float* zr  = z + (long long)wid * CC;
    float vals[3];
    float s = 0.f, s2 = 0.f;
    #pragma unroll
    for (int i = 0; i < 3; ++i) {
        float v = row[lane + 32 * i];
        vals[i] = v; s += v; s2 += v * v;
    }
    #pragma unroll
    for (int off = 16; off > 0; off >>= 1) {
        s  += __shfl_xor(s,  off, 32);
        s2 += __shfl_xor(s2, off, 32);
    }
    float mu  = s * (1.f / CC);
    float var = s2 * (1.f / CC) - mu * mu;
    float inv = rsqrtf(var + 1e-5f);
    #pragma unroll
    for (int i = 0; i < 3; ++i) {
        int   c   = lane + 32 * i;
        float zn  = zr[c];
        float sil = zn / (1.f + __expf(-zn));
        row[c] = ((vals[i] - mu) * inv * g[c] + bta[c]) * sil;
    }
}

// [BHW,C] -> NCHW output
__global__ void transpose_out_kernel(const float* __restrict__ y,
                                     float* __restrict__ out)
{
    long long idx = (long long)blockIdx.x * blockDim.x + threadIdx.x;
    if (idx >= (long long)BB * CC * HWN) return;
    int wcol = (int)(idx % WW);
    int h    = (int)((idx / WW) % HH);
    int c    = (int)((idx / (long long)HWN) % CC);
    int b    = (int)(idx / ((long long)CC * HWN));
    out[idx] = y[((long long)(b * HWN) + h * WW + wcol) * CC + c];
}

// ---------------------------------------------------------------------------
// Host-side orchestration
// ---------------------------------------------------------------------------
static inline long long cdiv(long long a, long long b) { return (a + b - 1) / b; }

static void launch_gemm(const float* A, long long lda, long long sA,
                        const float* Bm, long long ldb, long long sB, int opB,
                        float* Cm, long long ldc, long long sC,
                        const float* bias, int relu,
                        int M, int N, int K, int batch, hipStream_t s)
{
    dim3 grid((unsigned)(N / TN), (unsigned)(M / TM), (unsigned)batch);
    if (opB)
        gemm_bf16_wmma<1><<<grid, dim3(128, 1, 1), 0, s>>>(
            A, lda, sA, Bm, ldb, sB, Cm, ldc, sC, bias, relu, K);
    else
        gemm_bf16_wmma<0><<<grid, dim3(128, 1, 1), 0, s>>>(
            A, lda, sA, Bm, ldb, sB, Cm, ldc, sC, bias, relu, K);
}

extern "C" void kernel_launch(void* const* d_in, const int* in_sizes, int n_in,
                              void* d_out, int out_size, void* d_ws, size_t ws_size,
                              hipStream_t stream)
{
    (void)in_sizes; (void)n_in; (void)out_size; (void)ws_size;

    const float* x    = (const float*)d_in[0];   // [B,C,H,W]
    const float* fe   = (const float*)d_in[1];   // [H,W,C]
    const float* ot   = (const float*)d_in[2];   // [B,H,W,C]
    const float* dw_w = (const float*)d_in[3];   // [C,1,3,3]
    const float* dw_b = (const float*)d_in[4];
    const float* lw   = (const float*)d_in[5];   // [2C,C]
    const float* lb   = (const float*)d_in[6];
    const float* kw   = (const float*)d_in[7];   // [C,C]
    const float* kb   = (const float*)d_in[8];
    const float* l2w  = (const float*)d_in[9];   // [2C,2C]
    const float* l2b  = (const float*)d_in[10];
    const float* k2w  = (const float*)d_in[11];  // [2C,2C]
    const float* k2b  = (const float*)d_in[12];
    const float* l3w  = (const float*)d_in[13];  // [C,2C]
    const float* l3b  = (const float*)d_in[14];
    const float* lng  = (const float*)d_in[15];
    const float* lnb  = (const float*)d_in[16];
    const float* ow   = (const float*)d_in[17];  // [C,C]
    const float* ob   = (const float*)d_in[18];
    float* out = (float*)d_out;

    // ---- workspace layout (floats) ----
    float* ws = (float*)d_ws;
    long long off = 0;
    float* R1   = ws + off; off += (long long)BHW * C2;   // big slab A (2C)
    float* R3   = ws + off; off += (long long)BHW * C2;   // big slab B (2C)
    float* Rz   = ws + off; off += (long long)BHW * CC;   // z branch
    float* Rsum = ws + off; off += (long long)HWN * C2;   // batch-summed cat
    float* Rsk  = ws + off; off += (long long)HWN * C2;   // skfeat
    float* Rk   = ws + off; off += (long long)HWN * C2;   // k / k2
    float* Rfe  = ws + off; off += (long long)HWN * CC;   // fe + mean(ot)
    float* Rcn  = ws + off; off += (long long)HH * HH;
    float* RcnT = ws + off; off += (long long)HH * HH;
    float* Re   = ws + off; off += (long long)HH * HH;

    const long long NE_BHWC  = (long long)BHW * CC;
    const long long NE_BHW2C = (long long)BHW * C2;
    const long long sC1 = (long long)HWN * CC;   // per-batch stride, 96 ch
    const long long sC2 = (long long)HWN * C2;   // per-batch stride, 192 ch

    dim3 blk(256);

    // 0) precompute cos matrix + decay exponent map
    precompute_kernel<<<dim3((unsigned)cdiv(HH * HH, 256)), blk, 0, stream>>>(Rcn, RcnT, Re);

    // 1) depthwise conv NCHW -> NHWC into R1 (96-wide)
    dwconv_kernel<<<dim3((unsigned)cdiv(NE_BHWC, 256)), blk, 0, stream>>>(x, dw_w, dw_b, R1);

    // 2) lin1: x-part -> R3, z-part -> Rz
    launch_gemm(R1, CC, 0, lw,           CC, 0, 1, R3, CC, 0, lb,      0, BHW, CC, CC, 1, stream);
    launch_gemm(R1, CC, 0, lw + CC * CC, CC, 0, 1, Rz, CC, 0, lb + CC, 0, BHW, CC, CC, 1, stream);

    // 3) k = relu((fe + mean_b ot) @ k_w^T + k_b)
    meanfe_kernel<<<dim3((unsigned)cdiv((long long)HWN * CC, 256)), blk, 0, stream>>>(fe, ot, Rfe);
    launch_gemm(Rfe, CC, 0, kw, CC, 0, 1, Rk, CC, 0, kb, 1, HWN, CC, CC, 1, stream);

    // 4) heat pass 1: dct2 -> decay -> idct2  (R3 holds x, ping-pong with R1)
    launch_gemm(Rcn, HH, 0, R3, WW * CC, sC1, 0, R1, WW * CC, sC1, nullptr, 0, HH, WW * CC, HH, BB, stream);
    launch_gemm(Rcn, HH, 0, R1, CC, WW * CC, 0, R3, CC, WW * CC, nullptr, 0, WW, CC, WW, BB * HH, stream);
    decay_kernel<<<dim3((unsigned)cdiv(NE_BHWC, 256)), blk, 0, stream>>>(R3, Rk, Re, CC, NE_BHWC);
    launch_gemm(RcnT, HH, 0, R3, WW * CC, sC1, 0, R1, WW * CC, sC1, nullptr, 0, HH, WW * CC, HH, BB, stream);
    launch_gemm(RcnT, HH, 0, R1, CC, WW * CC, 0, R3, CC, WW * CC, nullptr, 0, WW, CC, WW, BB * HH, stream);

    // 5) concat [x | outline] -> R1 as [BHW,192]
    concat_kernel<<<dim3((unsigned)cdiv(NE_BHW2C, 256)), blk, 0, stream>>>(R3, ot, R1);

    // 6) skfeat = mean_b(cat) @ lin2^T + lin2_b ; k2 = relu(skfeat @ k2^T + k2_b)
    batchsum_kernel<<<dim3((unsigned)cdiv((long long)HWN * C2, 256)), blk, 0, stream>>>(R1, Rsum);
    launch_gemm(Rsum, C2, 0, l2w, C2, 0, 1, Rsk, C2, 0, l2b, 0, HWN, C2, C2, 1, stream);
    launch_gemm(Rsk,  C2, 0, k2w, C2, 0, 1, Rk,  C2, 0, k2b, 1, HWN, C2, C2, 1, stream);

    // 7) heat pass 2 on cat (192-wide), ping-pong R1 <-> R3
    launch_gemm(Rcn, HH, 0, R1, WW * C2, sC2, 0, R3, WW * C2, sC2, nullptr, 0, HH, WW * C2, HH, BB, stream);
    launch_gemm(Rcn, HH, 0, R3, C2, WW * C2, 0, R1, C2, WW * C2, nullptr, 0, WW, C2, WW, BB * HH, stream);
    decay_kernel<<<dim3((unsigned)cdiv(NE_BHW2C, 256)), blk, 0, stream>>>(R1, Rk, Re, C2, NE_BHW2C);
    launch_gemm(RcnT, HH, 0, R1, WW * C2, sC2, 0, R3, WW * C2, sC2, nullptr, 0, HH, WW * C2, HH, BB, stream);
    launch_gemm(RcnT, HH, 0, R3, C2, WW * C2, 0, R1, C2, WW * C2, nullptr, 0, WW, C2, WW, BB * HH, stream);

    // 8) lin3 (2C -> C) into R3
    launch_gemm(R1, C2, 0, l3w, C2, 0, 1, R3, CC, 0, l3b, 0, BHW, CC, C2, 1, stream);

    // 9) layernorm + silu(z) gate in place on R3
    ln_gate_kernel<<<dim3((unsigned)cdiv((long long)BHW * 32, 256)), blk, 0, stream>>>(
        R3, Rz, lng, lnb, BHW);

    // 10) out projection -> R1 (96-wide), then NHWC -> NCHW
    launch_gemm(R3, CC, 0, ow, CC, 0, 1, R1, CC, 0, ob, 0, BHW, CC, CC, 1, stream);
    transpose_out_kernel<<<dim3((unsigned)cdiv(NE_BHWC, 256)), blk, 0, stream>>>(R1, out);
}
